// E_GCL_65712999628899
// MI455X (gfx1250) — matrix operand order repaired and checked by reading, not compiled, not run on hardware
//
#include <hip/hip_runtime.h>
#include <math.h>

typedef __attribute__((ext_vector_type(16))) _Float16 v16h;
typedef __attribute__((ext_vector_type(8)))  _Float16 v8h;
typedef __attribute__((ext_vector_type(8)))  float    v8f;

#define Bb 2
#define Nn 384
#define Ff 64
#define Hh 128
#define Ee 4
#define KE 160        // einp K padded from 133 to 5*32
#define KEREAL 133
#define NW 4          // waves per edge-block

// ---- workspace layout (bytes) ----
#define OFF_WE1 0         // 5*8*512 halves  = 40960 B
#define OFF_WE2 40960     // 4*8*512 halves  = 32768 B
#define OFF_WA1 73728     // 32768 B
#define OFF_WC1 106496    // 32768 B
#define OFF_WN1 139264    // 8*8*512 halves  = 65536 B
#define OFF_WN2 204800    // 4*4*512 halves  = 16384 B
#define OFF_AGG 221184    // 768*128 f32     = 393216 B

// ---------------------------------------------------------------------------
// Weight swizzle: f32 row-major [K x Ncols] -> f16 B-fragment layout.
// Fragment (kc,nc): lane l, half h holds W[kc*32 + (h&7) + (h>>3)*16 + (l>=16?8:0), nc*16 + (l&15)]
// stored contiguously per lane so a B operand is one 32-byte load.
// ---------------------------------------------------------------------------
__global__ void swz_kernel(const float* __restrict__ src, _Float16* __restrict__ dst,
                           int Kreal, int KC, int Ncols) {
  int NC = Ncols >> 4;
  int total = KC * NC * 512;
  for (int e = blockIdx.x * blockDim.x + threadIdx.x; e < total;
       e += gridDim.x * blockDim.x) {
    int frag = e >> 9;
    int within = e & 511;
    int l = within >> 4, hh = within & 15;
    int kc = frag / NC, nc = frag % NC;
    int k = kc * 32 + (hh & 7) + ((hh >> 3) << 4) + ((l >= 16) ? 8 : 0);
    int n = nc * 16 + (l & 15);
    float v = (k < Kreal) ? src[(size_t)k * Ncols + n] : 0.0f;
    dst[e] = (_Float16)v;
  }
}

// ---------------------------------------------------------------------------
// Fragment loaders
// ---------------------------------------------------------------------------
__device__ inline v16h load_a_frag(const _Float16* buf, int ld, int kchunk, int lane) {
  int m  = lane & 15;
  int kb = kchunk + ((lane & 16) ? 8 : 0);
  const _Float16* p = buf + m * ld + kb;
  v8h lo = *(const v8h*)(p);        // K = kb .. kb+7
  v8h hi = *(const v8h*)(p + 16);   // K = kb+16 .. kb+23
  v16h a;
#pragma unroll
  for (int t = 0; t < 8; ++t) { a[t] = lo[t]; a[8 + t] = hi[t]; }
  return a;
}

__device__ inline v16h load_b_frag(const _Float16* w, int frag, int lane) {
  return *(const v16h*)(w + ((size_t)frag * 32 + lane) * 16);
}

// Opaque zero (in SGPR): blocks LICM/CSE of weight-fragment loads across the
// tile loop while keeping the pointer kernarg-derived -> global_load (not flat).
__device__ inline int opaque_zero() {
  int v = 0;
  asm volatile("" : "+s"(v));
  return v;
}

// D = A(16xK from LDS) * W(KxNC*16, pre-swizzled) ; f32 accum.
// B fragments are software-pipelined with prefetch distance 2.
template <int KC, int NC>
__device__ inline void gemm16(const _Float16* Ab, int lda, const _Float16* W, v8f* D) {
  constexpr int T = KC * NC;
  int lane = threadIdx.x & 31;
  const _Float16* Wo = W + opaque_zero();
  v16h a[KC];
#pragma unroll
  for (int kc = 0; kc < KC; ++kc) a[kc] = load_a_frag(Ab, lda, kc * 32, lane);

  v16h bf[4];
  bf[0] = load_b_frag(Wo, 0 * NC + 0, lane);  // t=0: kc=0,nc=0
  bf[1] = load_b_frag(Wo, 1 * NC + 0, lane);  // t=1: kc=1,nc=0
  v8f zero = {};
  v8f d = zero;
#pragma unroll
  for (int t = 0; t < T; ++t) {
    const int nc = t / KC, kc = t % KC;
    if (t + 2 < T) {
      const int t2 = t + 2;
      bf[t2 & 3] = load_b_frag(Wo, (t2 % KC) * NC + (t2 / KC), lane);
    }
    d = __builtin_amdgcn_wmma_f32_16x16x32_f16(false, a[kc], false, bf[t & 3],
                                               (short)0, d, false, false);
    if (kc == KC - 1) { D[nc] = d; d = zero; }
    // Scheduling fence: bounds live B fragments to the 4-slot pipeline.
    asm volatile("" ::: "memory");
  }
}

// ---------------------------------------------------------------------------
// Edge kernel: one block per (b,i); 4 waves x 6 tiles of 16 j-rows.
// Produces edge_feat (f32), per-node agg (ws), coord_out.
// ---------------------------------------------------------------------------
__global__ __launch_bounds__(NW * 32) void edge_kernel(
    const float* __restrict__ h, const float* __restrict__ coord,
    const float* __restrict__ eattr, const float* __restrict__ mask,
    const _Float16* __restrict__ We1s, const float* __restrict__ be1,
    const _Float16* __restrict__ We2s, const float* __restrict__ be2,
    const _Float16* __restrict__ Wa1s, const float* __restrict__ ba1,
    const float* __restrict__ Wa2, const float* __restrict__ ba2,
    const _Float16* __restrict__ Wc1s, const float* __restrict__ bc1,
    const float* __restrict__ Wc2,
    float* __restrict__ ef_out, float* __restrict__ coord_out,
    float* __restrict__ agg_out) {
  __shared__ __align__(16) _Float16 P[NW][16 * KE];  // einp / m2 staging
  __shared__ __align__(16) _Float16 Q[NW][16 * Hh];  // m1 / q / u staging
  __shared__ _Float16 hi_h[Ff];
  __shared__ float agg[Hh];
  __shared__ float csum[3];
  __shared__ float msum_s;
  __shared__ float pbuf[NW][32];
  __shared__ float attn[NW][16];
  __shared__ float cdf[NW][16][3];
  __shared__ float rad[NW][16];
  __shared__ float mjb[NW][16];

  const int blk = blockIdx.x;
  const int b = blk / Nn, i = blk % Nn;
  const int wave = threadIdx.x >> 5, lane = threadIdx.x & 31;

  const float mi  = mask[b * Nn + i];
  const float cix = coord[(b * Nn + i) * 3 + 0] * mi;
  const float ciy = coord[(b * Nn + i) * 3 + 1] * mi;
  const float ciz = coord[(b * Nn + i) * 3 + 2] * mi;

  if (threadIdx.x < Hh) agg[threadIdx.x] = 0.0f;
  if (threadIdx.x < Ff) hi_h[threadIdx.x] = (_Float16)(h[(b * Nn + i) * Ff + threadIdx.x] * mi);
  if (threadIdx.x < 3) csum[threadIdx.x] = 0.0f;
  if (wave == 0) {  // sum of mask over j for denom
    float s = 0.0f;
    for (int j = lane; j < Nn; j += 32) s += mask[b * Nn + j];
    pbuf[0][lane] = s;
    float t = 0.0f;
    if (lane == 0) {
      for (int k = 0; k < 32; ++k) t += pbuf[0][k];
      msum_s = t;
    }
  }
  __syncthreads();

  _Float16* Pw = P[wave];
  _Float16* Qw = Q[wave];
  const int ncol = lane & 15, mhalf = (lane >> 4) << 3;
  const size_t ea_base = (size_t)(b * Nn + i) * Nn * Ee;
  v8f D[8];

  for (int tile = wave; tile < Nn / 16; tile += NW) {
    const int j0 = tile * 16;
    // --- per-row geometry ---
    if (lane < 16) {
      int j = j0 + lane;
      float mj = mask[b * Nn + j];
      float m2 = mi * mj;
      float cjx = coord[(b * Nn + j) * 3 + 0] * mj;
      float cjy = coord[(b * Nn + j) * 3 + 1] * mj;
      float cjz = coord[(b * Nn + j) * 3 + 2] * mj;
      float dx = cix - cjx, dy = ciy - cjy, dz = ciz - cjz;
      rad[wave][lane] = (dx * dx + dy * dy + dz * dz) * m2;
      cdf[wave][lane][0] = dx * m2;
      cdf[wave][lane][1] = dy * m2;
      cdf[wave][lane][2] = dz * m2;
      mjb[wave][lane] = mj;
    }
    // --- build einp tile (16 x 160): [h_j | h_i | radial | edge_attr | 0-pad] ---
    for (int idx = lane; idx < 16 * KE; idx += 32) {
      int row = idx / KE, c = idx % KE;
      int j = j0 + row;
      float v;
      if (c < Ff)             v = h[(b * Nn + j) * Ff + c] * mjb[wave][row];
      else if (c < 2 * Ff)    v = (float)hi_h[c - Ff];
      else if (c == 2 * Ff)   v = rad[wave][row];
      else if (c < KEREAL)    v = eattr[ea_base + (size_t)j * Ee + (c - 129)];
      else                    v = 0.0f;
      Pw[row * KE + c] = (_Float16)v;
    }
    __syncthreads();

    // --- GEMM1: m1 = tanh(einp @ We1 + be1) -> Q ---
    gemm16<5, 8>(Pw, KE, We1s, D);
#pragma unroll
    for (int nc = 0; nc < 8; ++nc) {
      int col = nc * 16 + ncol;
      float bv = be1[col];
#pragma unroll
      for (int r = 0; r < 8; ++r)
        Qw[(r + mhalf) * Hh + col] = (_Float16)tanhf(D[nc][r] + bv);
    }
    __syncthreads();

    // --- GEMM2: m2 = tanh(m1 @ We2 + be2) -> P (stride KE) ---
    gemm16<4, 8>(Qw, Hh, We2s, D);
#pragma unroll
    for (int nc = 0; nc < 8; ++nc) {
      int col = nc * 16 + ncol;
      float bv = be2[col];
#pragma unroll
      for (int r = 0; r < 8; ++r)
        Pw[(r + mhalf) * KE + col] = (_Float16)tanhf(D[nc][r] + bv);
    }
    __syncthreads();

    // --- GEMM3: q = tanh(m2 @ Wa1 + ba1) -> Q ---
    gemm16<4, 8>(Pw, KE, Wa1s, D);
#pragma unroll
    for (int nc = 0; nc < 8; ++nc) {
      int col = nc * 16 + ncol;
      float bv = ba1[col];
#pragma unroll
      for (int r = 0; r < 8; ++r)
        Qw[(r + mhalf) * Hh + col] = (_Float16)tanhf(D[nc][r] + bv);
    }
    __syncthreads();

    // --- attn = sigmoid(q @ Wa2 + ba2), per row ---
    {
      int row = lane & 15, cb = (lane >> 4) * 64;
      float s = 0.0f;
      for (int c = 0; c < 64; ++c) s += (float)Qw[row * Hh + cb + c] * Wa2[cb + c];
      pbuf[wave][lane] = s;
    }
    __syncthreads();
    if (lane < 16) {
      float z = pbuf[wave][lane] + pbuf[wave][lane + 16] + ba2[0];
      attn[wave][lane] = 1.0f / (1.0f + expf(-z));
    }
    __syncthreads();

    // --- edge_feat = m2 * attn : write f32 + accumulate agg ---
    {
      size_t base = ((size_t)(b * Nn + i) * Nn + j0) * Hh;
#pragma unroll
      for (int cc = 0; cc < 4; ++cc) {
        int col = lane + cc * 32;
        float s = 0.0f;
        for (int row = 0; row < 16; ++row) {
          float v = (float)Pw[row * KE + col] * attn[wave][row];
          ef_out[base + (size_t)row * Hh + col] = v;
          s += v;
        }
        atomicAdd(&agg[col], s);
      }
    }

    // --- GEMM4: u = tanh(attn*(m2 @ Wc1) + bc1) -> Q  (ef@Wc1 == diag(attn)*(m2@Wc1)) ---
    gemm16<4, 8>(Pw, KE, Wc1s, D);
#pragma unroll
    for (int nc = 0; nc < 8; ++nc) {
      int col = nc * 16 + ncol;
      float bv = bc1[col];
#pragma unroll
      for (int r = 0; r < 8; ++r) {
        int row = r + mhalf;
        Qw[row * Hh + col] = (_Float16)tanhf(attn[wave][row] * D[nc][r] + bv);
      }
    }
    __syncthreads();

    // --- embed = u @ Wc2 ; accumulate coord sums ---
    {
      int row = lane & 15, cb = (lane >> 4) * 64;
      float s = 0.0f;
      for (int c = 0; c < 64; ++c) s += (float)Qw[row * Hh + cb + c] * Wc2[cb + c];
      pbuf[wave][lane] = s;
    }
    __syncthreads();
    if (lane < 16) {
      float e = pbuf[wave][lane] + pbuf[wave][lane + 16];
      atomicAdd(&csum[0], cdf[wave][lane][0] * e);
      atomicAdd(&csum[1], cdf[wave][lane][1] * e);
      atomicAdd(&csum[2], cdf[wave][lane][2] * e);
    }
    __syncthreads();
  }

  __syncthreads();
  if (threadIdx.x < 3) {
    float denom = mi * msum_s + 1e-10f;
    float c0 = (threadIdx.x == 0) ? cix : (threadIdx.x == 1 ? ciy : ciz);
    coord_out[(b * Nn + i) * 3 + threadIdx.x] = (c0 + csum[threadIdx.x] / denom) * mi;
  }
  if (threadIdx.x < Hh)
    agg_out[(size_t)(b * Nn + i) * Hh + threadIdx.x] = agg[threadIdx.x];
}

// ---------------------------------------------------------------------------
// Node kernel: 16 nodes per block (1 wave). ninp = [h | node_attr | agg] (K=256)
// ---------------------------------------------------------------------------
__global__ __launch_bounds__(32) void node_kernel(
    const float* __restrict__ h, const float* __restrict__ nattr,
    const float* __restrict__ mask, const float* __restrict__ agg_in,
    const _Float16* __restrict__ Wn1s, const float* __restrict__ bn1,
    const _Float16* __restrict__ Wn2s, const float* __restrict__ bn2,
    float* __restrict__ hout) {
  __shared__ __align__(16) _Float16 P[16 * 256];
  __shared__ __align__(16) _Float16 Q[16 * Hh];
  const int node0 = blockIdx.x * 16;
  const int lane = threadIdx.x;

  for (int idx = lane; idx < 16 * 256; idx += 32) {
    int row = idx >> 8, c = idx & 255;
    int node = node0 + row;
    float m = mask[node];
    float v;
    if (c < Ff)            v = h[node * Ff + c] * m;
    else if (c < 2 * Ff)   v = nattr[node * Ff + (c - Ff)] * m;
    else                   v = agg_in[(size_t)node * Hh + (c - 2 * Ff)] * m;
    P[idx] = (_Float16)v;
  }
  __syncthreads();

  v8f D[8];
  gemm16<8, 8>(P, 256, Wn1s, D);
  const int ncol = lane & 15, mhalf = (lane >> 4) << 3;
#pragma unroll
  for (int nc = 0; nc < 8; ++nc) {
    int col = nc * 16 + ncol;
    float bv = bn1[col];
#pragma unroll
    for (int r = 0; r < 8; ++r)
      Q[(r + mhalf) * Hh + col] = (_Float16)tanhf(D[nc][r] + bv);
  }
  __syncthreads();

  v8f D2[4];
  gemm16<4, 4>(Q, Hh, Wn2s, D2);
#pragma unroll
  for (int nc = 0; nc < 4; ++nc) {
    int col = nc * 16 + ncol;
    float bv = bn2[col];
#pragma unroll
    for (int r = 0; r < 8; ++r) {
      int node = node0 + r + mhalf;
      float m = mask[node];
      hout[node * Ff + col] = (h[node * Ff + col] * m + D2[nc][r] + bv) * m;
    }
  }
}

// ---------------------------------------------------------------------------
extern "C" void kernel_launch(void* const* d_in, const int* in_sizes, int n_in,
                              void* d_out, int out_size, void* d_ws, size_t ws_size,
                              hipStream_t stream) {
  (void)in_sizes; (void)n_in; (void)out_size; (void)ws_size;
  const float* h     = (const float*)d_in[0];
  const float* coord = (const float*)d_in[1];
  const float* eattr = (const float*)d_in[2];
  const float* nattr = (const float*)d_in[3];
  const float* mask  = (const float*)d_in[4];
  const float* We1 = (const float*)d_in[5];  const float* be1 = (const float*)d_in[6];
  const float* We2 = (const float*)d_in[7];  const float* be2 = (const float*)d_in[8];
  const float* Wa1 = (const float*)d_in[9];  const float* ba1 = (const float*)d_in[10];
  const float* Wa2 = (const float*)d_in[11]; const float* ba2 = (const float*)d_in[12];
  const float* Wn1 = (const float*)d_in[13]; const float* bn1 = (const float*)d_in[14];
  const float* Wn2 = (const float*)d_in[15]; const float* bn2 = (const float*)d_in[16];
  const float* Wc1 = (const float*)d_in[17]; const float* bc1 = (const float*)d_in[18];
  const float* Wc2 = (const float*)d_in[19];

  char* ws = (char*)d_ws;
  _Float16* We1s = (_Float16*)(ws + OFF_WE1);
  _Float16* We2s = (_Float16*)(ws + OFF_WE2);
  _Float16* Wa1s = (_Float16*)(ws + OFF_WA1);
  _Float16* Wc1s = (_Float16*)(ws + OFF_WC1);
  _Float16* Wn1s = (_Float16*)(ws + OFF_WN1);
  _Float16* Wn2s = (_Float16*)(ws + OFF_WN2);
  float* aggw = (float*)(ws + OFF_AGG);

  float* hout  = (float*)d_out;                 // [B,N,F]
  float* cout  = hout + Bb * Nn * Ff;           // [B,N,3]
  float* efout = cout + Bb * Nn * 3;            // [B,N,N,H]

  swz_kernel<<<64, 256, 0, stream>>>(We1, We1s, KEREAL, 5, Hh);
  swz_kernel<<<64, 256, 0, stream>>>(We2, We2s, Hh, 4, Hh);
  swz_kernel<<<64, 256, 0, stream>>>(Wa1, Wa1s, Hh, 4, Hh);
  swz_kernel<<<64, 256, 0, stream>>>(Wc1, Wc1s, Hh, 4, Hh);
  swz_kernel<<<64, 256, 0, stream>>>(Wn1, Wn1s, 256, 8, Hh);
  swz_kernel<<<64, 256, 0, stream>>>(Wn2, Wn2s, Hh, 4, Ff);

  edge_kernel<<<Bb * Nn, NW * 32, 0, stream>>>(
      h, coord, eattr, mask, We1s, be1, We2s, be2, Wa1s, ba1, Wa2, ba2,
      Wc1s, bc1, Wc2, efout, cout, aggw);

  node_kernel<<<(Bb * Nn) / 16, 32, 0, stream>>>(
      h, nattr, mask, aggw, Wn1s, bn1, Wn2s, bn2, hout);
}
